// EntmaxAttention_26388279067318
// MI455X (gfx1250) — compile-verified, matrix operand order checked by reading, not run on hardware
//
#include <hip/hip_runtime.h>
#include <hip/hip_bf16.h>
#include <stdint.h>

typedef __bf16 bf16;
typedef __attribute__((ext_vector_type(16))) __bf16 v16bf;
typedef __attribute__((ext_vector_type(8)))  float  v8f;
typedef __attribute__((ext_vector_type(4)))  float  v4f;

#define B_  8
#define N_  1024
#define C_  1024
#define H_  8
#define D_  128

__device__ __forceinline__ v8f wmma_bf16(v16bf a, v16bf b, v8f c) {
  return __builtin_amdgcn_wmma_f32_16x16x32_bf16(false, a, false, b, (short)0, c, false, false);
}

__device__ __forceinline__ float wave_sum(float v) {
#pragma unroll
  for (int m = 16; m >= 1; m >>= 1) v += __shfl_xor(v, m, 32);
  return v;
}
__device__ __forceinline__ float wave_max(float v) {
#pragma unroll
  for (int m = 16; m >= 1; m >>= 1) v = fmaxf(v, __shfl_xor(v, m, 32));
  return v;
}

// ---------------- converters ----------------
__global__ __launch_bounds__(256) void cvt_f32_bf16(const float* __restrict__ src,
                                                    bf16* __restrict__ dst, int n) {
  int i = (blockIdx.x * 256 + threadIdx.x) * 4;
  if (i < n) {
#pragma unroll
    for (int j = 0; j < 4; ++j) dst[i + j] = (bf16)src[i + j];
  }
}

// W[1024x1024] f32 -> Wt[1024x1024] bf16 (transposed: Wt[n][k] = W[k][n])
__global__ __launch_bounds__(256) void tcvt_f32_bf16(const float* __restrict__ src,
                                                     bf16* __restrict__ dst) {
  int idx = blockIdx.x * 256 + threadIdx.x;
  int i = idx >> 10, j = idx & 1023;
  dst[(size_t)j * 1024 + i] = (bf16)src[idx];
}

// ---------------- WMMA GEMM: Y[M,Nc] = A[M,K] * Bt[Nc,K]^T + bias ----------------
// MODE 0: write bf16 row-major. MODE 1: write f32 row-major.
// MODE 2: write bf16 V-transposed: Vt[((b*8+h)*128 + d)*1024 + n]
template <int MODE>
__global__ __launch_bounds__(256) void gemm_wmma(const bf16* __restrict__ A,
                                                 const bf16* __restrict__ Bt,
                                                 const float* __restrict__ bias,
                                                 void* __restrict__ Out,
                                                 int M, int Nc, int K) {
  const int lane = threadIdx.x & 31;
  const int wave = threadIdx.x >> 5;
  const int rowBase = blockIdx.x * 32 + (wave & 1) * 16;
  const int colBase = blockIdx.y * 256 + (wave >> 1) * 64;
  const int m  = lane & 15;
  const int ho = (lane >> 4) * 16;   // K-half select: contiguous k = 16h + s, same order for A and B

  v8f acc[4] = {};
  for (int kk = 0; kk < K; kk += 32) {
    const bf16* ap = A + (size_t)(rowBase + m) * K + kk + ho;
    v16bf a = *(const v16bf*)ap;
    if (kk + 64 <= K) __builtin_prefetch(ap + 64, 0, 0);   // global_prefetch_b8
#pragma unroll
    for (int t = 0; t < 4; ++t) {
      const bf16* bp = Bt + (size_t)(colBase + t * 16 + m) * K + kk + ho;
      v16bf b = *(const v16bf*)bp;
      acc[t] = wmma_bf16(a, b, acc[t]);
    }
  }

#pragma unroll
  for (int t = 0; t < 4; ++t) {
#pragma unroll
    for (int i = 0; i < 8; ++i) {
      int row = rowBase + i + ((lane >> 4) * 8);   // C/D layout: vgpr i -> M=i / i+8
      int col = colBase + t * 16 + (lane & 15);
      float val = acc[t][i] + bias[col];
      if (MODE == 0) {
        ((bf16*)Out)[(size_t)row * Nc + col] = (bf16)val;
      } else if (MODE == 1) {
        ((float*)Out)[(size_t)row * Nc + col] = val;
      } else {
        int bb = row >> 10, nn = row & 1023;
        int hh = col >> 7,  dd = col & 127;
        ((bf16*)Out)[(((size_t)(bb * 8 + hh) * 128 + dd) << 10) + nn] = (bf16)val;
      }
    }
  }
}

// ---------------- fused QK^T -> entmax(1.5, 50-iter bisect) -> attn*V ----------------
// Block: 256 threads (8 waves), handles one (b,h) and a 16-row tile of Q.
__global__ __launch_bounds__(256) void entmax_attn(const bf16* __restrict__ Q,
                                                   const bf16* __restrict__ Kb,
                                                   const bf16* __restrict__ Vt,
                                                   bf16* __restrict__ ctx,
                                                   float* __restrict__ attnOut) {
  const int lane = threadIdx.x & 31;
  const int wave = threadIdx.x >> 5;
  const int blk  = blockIdx.x;            // B*H*(N/16) = 4096
  const int rtile = blk & 63;
  const int bh    = blk >> 6;
  const int b = bh >> 3, h = bh & 7;
  const int rowBase = rtile * 16;

  __shared__ __align__(64) float sc[16 * 1024];   // 64 KB score tile

  const int m  = lane & 15;
  const int ho = (lane >> 4) * 16;

  // xm = (alpha-1) * qk / sqrt(D) = 0.5/sqrt(128) * qk
  const float xscale = 0.5f * 0.08838834764831845f;

  // --- load 16x128 Q fragments once (4 K-steps of 32) ---
  const bf16* qp = Q + ((size_t)(b * N_) + rowBase + m) * C_ + h * D_;
  v16bf aq[4];
#pragma unroll
  for (int ks = 0; ks < 4; ++ks) aq[ks] = *(const v16bf*)(qp + ks * 32 + ho);

  // --- scores: wave owns columns [wave*128, wave*128+128) ---
  for (int t = 0; t < 8; ++t) {
    const int colBase = wave * 128 + t * 16;
    v8f acc = {};
#pragma unroll
    for (int ks = 0; ks < 4; ++ks) {
      v16bf bk = *(const v16bf*)(Kb + ((size_t)(b * N_) + colBase + m) * C_ + h * D_ + ks * 32 + ho);
      acc = wmma_bf16(aq[ks], bk, acc);
    }
#pragma unroll
    for (int i = 0; i < 8; ++i) {
      int rr = i + ((lane >> 4) * 8);
      int cc = colBase + (lane & 15);
      sc[rr * 1024 + cc] = acc[i] * xscale;
    }
  }
  __syncthreads();

  // --- pull rows into registers: wave handles rows 2*wave, 2*wave+1 ---
  float xr[2][32];
#pragma unroll
  for (int r = 0; r < 2; ++r) {
    const float* rowp = &sc[(2 * wave + r) * 1024 + lane * 32];
#pragma unroll
    for (int j = 0; j < 8; ++j) {
      v4f v = *(const v4f*)(rowp + j * 4);
      xr[r][j * 4 + 0] = v[0]; xr[r][j * 4 + 1] = v[1];
      xr[r][j * 4 + 2] = v[2]; xr[r][j * 4 + 3] = v[3];
    }
  }
  __syncthreads();   // rows captured -> LDS reusable as bf16 attn tile

  bf16* atn = (bf16*)sc;

#pragma unroll
  for (int r = 0; r < 2; ++r) {
    const int row = 2 * wave + r;
    // max over row
    float mx = -__builtin_inff();
#pragma unroll
    for (int j = 0; j < 32; ++j) mx = fmaxf(mx, xr[r][j]);
    mx = wave_max(mx);

    float tau_lo = mx - 1.0f;            // gp(1, 1.5) = 1
    float dm     = 0.96875f;             // tau_hi - tau_lo = 1 - (1/1024)^0.5
    // f_lo
    float s = 0.0f;
#pragma unroll
    for (int j = 0; j < 32; ++j) {
      float t = fmaxf(xr[r][j] - tau_lo, 0.0f);
      s += t * t;
    }
    const float f_lo = wave_sum(s) - 1.0f;

    float tau_m = tau_lo;
    for (int it = 0; it < 50; ++it) {
      dm *= 0.5f;
      tau_m = tau_lo + dm;
      float sm = 0.0f;
#pragma unroll
      for (int j = 0; j < 32; ++j) {
        float t = fmaxf(xr[r][j] - tau_m, 0.0f);
        sm += t * t;
      }
      sm = wave_sum(sm);
      float f_m = sm - 1.0f;
      if (f_m * f_lo >= 0.0f) tau_lo = tau_m;
    }

    // p = relu(xm - tau)^2, normalized
    float p[32];
    float ps = 0.0f;
#pragma unroll
    for (int j = 0; j < 32; ++j) {
      float t = fmaxf(xr[r][j] - tau_m, 0.0f);
      p[j] = t * t;
      ps += p[j];
    }
    ps = wave_sum(ps);
    const float inv = 1.0f / ps;

    float* arow = attnOut + (((size_t)bh * N_) + rowBase + row) * N_ + lane * 32;
#pragma unroll
    for (int j = 0; j < 8; ++j) {
      v4f pv;
      pv[0] = p[j * 4 + 0] * inv; pv[1] = p[j * 4 + 1] * inv;
      pv[2] = p[j * 4 + 2] * inv; pv[3] = p[j * 4 + 3] * inv;
      *(v4f*)(arow + j * 4) = pv;
    }
#pragma unroll
    for (int j = 0; j < 32; ++j)
      atn[row * 1024 + lane * 32 + j] = (bf16)(p[j] * inv);
  }
  __syncthreads();

  // --- ctx = attn(16x1024) @ V(1024x128): wave owns ctx cols [wave*16, wave*16+16) ---
  v8f cacc = {};
  for (int m0 = 0; m0 < 1024; m0 += 32) {
    v16bf a  = *(const v16bf*)(atn + (size_t)(lane & 15) * 1024 + m0 + ho);           // LDS
    v16bf bv = *(const v16bf*)(Vt + (((size_t)bh * D_) + wave * 16 + (lane & 15)) * 1024 + m0 + ho);
    cacc = wmma_bf16(a, bv, cacc);
  }
#pragma unroll
  for (int i = 0; i < 8; ++i) {
    int rr = i + ((lane >> 4) * 8);
    int cc = wave * 16 + (lane & 15);
    ctx[((size_t)(b * N_) + rowBase + rr) * C_ + h * D_ + cc] = (bf16)cacc[i];
  }
}

// ---------------- host launch ----------------
extern "C" void kernel_launch(void* const* d_in, const int* in_sizes, int n_in,
                              void* d_out, int out_size, void* d_ws, size_t ws_size,
                              hipStream_t stream) {
  const float* x  = (const float*)d_in[0];
  const float* Wq = (const float*)d_in[1];
  const float* bq = (const float*)d_in[2];
  const float* Wk = (const float*)d_in[3];
  const float* bk = (const float*)d_in[4];
  const float* Wv = (const float*)d_in[5];
  const float* bv = (const float*)d_in[6];
  const float* Wo = (const float*)d_in[7];
  const float* bo = (const float*)d_in[8];

  const size_t BN = (size_t)B_ * N_;        // 8192
  const size_t XN = BN * C_;                // 8M elements

  uint8_t* ws = (uint8_t*)d_ws;
  size_t off = 0;
  bf16* xh  = (bf16*)(ws + off); off += XN * 2;                 // 16 MB
  bf16* Wqt = (bf16*)(ws + off); off += (size_t)C_ * C_ * 2;    // 2 MB
  bf16* Wkt = (bf16*)(ws + off); off += (size_t)C_ * C_ * 2;
  bf16* Wvt = (bf16*)(ws + off); off += (size_t)C_ * C_ * 2;
  bf16* Wot = (bf16*)(ws + off); off += (size_t)C_ * C_ * 2;
  bf16* Qb  = (bf16*)(ws + off); off += XN * 2;
  bf16* Kb  = (bf16*)(ws + off); off += XN * 2;
  bf16* Vt  = (bf16*)(ws + off); off += XN * 2;                 // [B*H][128][1024]
  bf16* ctx = (bf16*)(ws + off); off += XN * 2;

  float* outp    = (float*)d_out;                               // [B,N,C]
  float* attnOut = outp + XN;                                   // [B,H,N,N]

  // 1) convert inputs to bf16 (weights transposed for contiguous B-fragments)
  cvt_f32_bf16<<<(int)(XN / (256 * 4)), 256, 0, stream>>>(x, xh, (int)XN);
  tcvt_f32_bf16<<<4096, 256, 0, stream>>>(Wq, Wqt);
  tcvt_f32_bf16<<<4096, 256, 0, stream>>>(Wk, Wkt);
  tcvt_f32_bf16<<<4096, 256, 0, stream>>>(Wv, Wvt);
  tcvt_f32_bf16<<<4096, 256, 0, stream>>>(Wo, Wot);

  // 2) projections: Q, K row-major bf16; V written head-transposed
  dim3 ggrid(BN / 32, C_ / 256);
  gemm_wmma<0><<<ggrid, 256, 0, stream>>>(xh, Wqt, bq, (void*)Qb, (int)BN, C_, C_);
  gemm_wmma<0><<<ggrid, 256, 0, stream>>>(xh, Wkt, bk, (void*)Kb, (int)BN, C_, C_);
  gemm_wmma<2><<<ggrid, 256, 0, stream>>>(xh, Wvt, bv, (void*)Vt, (int)BN, C_, C_);

  // 3) fused QK^T -> entmax-1.5 bisection -> attn*V
  entmax_attn<<<B_ * H_ * (N_ / 16), 256, 0, stream>>>(Qb, Kb, Vt, ctx, attnOut);

  // 4) out = ctx @ Wo + bo (f32)
  gemm_wmma<1><<<ggrid, 256, 0, stream>>>(ctx, Wot, bo, (void*)outp, (int)BN, C_, C_);
}
